// GFRLayer_72009421685071
// MI455X (gfx1250) — compile-verified, matrix operand order checked by reading, not run on hardware
//
#include <hip/hip_runtime.h>
#include <math.h>

// ---------------- problem constants ----------------
#define Bdim 4
#define Tdim 2048
#define Ddim 2048
#define Kdim 4
#define Mdim (Bdim * Tdim)      // 8192 rows of z
#define LCHUNK 128              // scan chunk length
#define NCHUNK (Tdim / LCHUNK)  // 16 chunks
#define LN_EPS 1e-5f

// ---------------- CDNA5 WMMA types ----------------
typedef __attribute__((ext_vector_type(16))) __bf16 v16bf;
typedef __attribute__((ext_vector_type(8)))  float  v8f;
typedef __attribute__((ext_vector_type(4)))  int    v4i_t;

struct __align__(16) U4 { unsigned int x, y, z, w; };
union FragBF { U4 q[2]; v16bf v; };
struct __align__(8) US4 { unsigned short a, b, c, d; };

__device__ __forceinline__ unsigned short f32_to_bf16_rne(float f) {
    unsigned int u = __float_as_uint(f);
    unsigned int r = u + 0x7FFFu + ((u >> 16) & 1u);
    return (unsigned short)(r >> 16);
}
__device__ __forceinline__ float bf16_bits_to_f32(unsigned short h) {
    return __uint_as_float(((unsigned int)h) << 16);
}

// ---------------- gfx1250 async global->LDS path ----------------
#if __has_builtin(__builtin_amdgcn_global_load_async_to_lds_b128)
#define ASYNC_PATH 1
#else
#define ASYNC_PATH 0
#endif

__device__ __forceinline__ void copy16_g2l(const void* g, void* l) {
#if ASYNC_PATH
    __builtin_amdgcn_global_load_async_to_lds_b128(
        (__attribute__((address_space(1))) v4i_t*)(v4i_t*)(void*)g,
        (__attribute__((address_space(3))) v4i_t*)(v4i_t*)l,
        0, 0);
#else
    *(U4*)l = *(const U4*)g;   // synchronous fallback
#endif
}

template <int N>
__device__ __forceinline__ void wait_async_le() {
#if ASYNC_PATH
#if __has_builtin(__builtin_amdgcn_s_wait_asynccnt)
    __builtin_amdgcn_s_wait_asynccnt((short)N);
#else
    asm volatile("s_wait_asynccnt %0" :: "n"(N) : "memory");
#endif
#endif
}

// =====================================================================
// Kernel 0: split fp32 -> bf16 (hi, lo) planes, vectorized 4-wide
// =====================================================================
__global__ void __launch_bounds__(256)
split_bf16(const float* __restrict__ src, unsigned short* __restrict__ hi,
           unsigned short* __restrict__ lo, int n4) {
    int i = blockIdx.x * 256 + threadIdx.x;
    if (i >= n4) return;
    float4 v = ((const float4*)src)[i];
    US4 h, l;
    h.a = f32_to_bf16_rne(v.x); l.a = f32_to_bf16_rne(v.x - bf16_bits_to_f32(h.a));
    h.b = f32_to_bf16_rne(v.y); l.b = f32_to_bf16_rne(v.y - bf16_bits_to_f32(h.b));
    h.c = f32_to_bf16_rne(v.z); l.c = f32_to_bf16_rne(v.z - bf16_bits_to_f32(h.c));
    h.d = f32_to_bf16_rne(v.w); l.d = f32_to_bf16_rne(v.w - bf16_bits_to_f32(h.d));
    ((US4*)hi)[i] = h;
    ((US4*)lo)[i] = l;
}

// =====================================================================
// Kernel 1: z = x @ W^T  (split-bf16, 3x v_wmma_f32_16x16x32_bf16)
//   128x128 block tile, 8 waves, K step 32, double-buffered LDS fed by
//   async global->LDS b128 copies (ASYNCcnt pipelined one slab ahead).
// =====================================================================
#define LDA 40   // padded LDS row stride (bf16 elems) to spread banks

__global__ void __launch_bounds__(256)
gemm_split_bf16(const unsigned short* __restrict__ xh,
                const unsigned short* __restrict__ xl,
                const unsigned short* __restrict__ wh,
                const unsigned short* __restrict__ wl,
                float* __restrict__ z) {
    // [buffer][tile: 0=Ah 1=Al 2=Bh 3=Bl][128 rows * LDA]
    __shared__ __align__(16) unsigned short lds[2][4][128 * LDA];

    const int tid  = threadIdx.x;
    const int lane = tid & 31;
    const int wave = tid >> 5;        // 0..7
    const int wm   = wave >> 2;       // 0..1  (M direction, 64 rows each)
    const int wn   = wave & 3;        // 0..3  (N direction, 32 cols each)

    const int m0 = blockIdx.y * 128;
    const int n0 = blockIdx.x * 128;

    const int mrow = lane & 15;
    const int kh   = lane >> 4;       // K-half selector per WMMA A/B layout

    v8f acc[4][2] = {};

    // issue one 128x32 slab (4 tiles) : 8 async b128 per thread
    auto issue = [&](int buf, int k0) {
#pragma unroll
        for (int u = 0; u < 2; ++u) {
            int cid = tid + u * 256;          // 0..511 chunk id
            int row = cid >> 2;               // 0..127
            int cq  = (cid & 3) * 8;          // bf16 element offset in row
            size_t ga = (size_t)(m0 + row) * Ddim + k0 + cq;
            size_t gb = (size_t)(n0 + row) * Ddim + k0 + cq;
            int lo = row * LDA + cq;
            copy16_g2l(xh + ga, &lds[buf][0][lo]);
            copy16_g2l(xl + ga, &lds[buf][1][lo]);
            copy16_g2l(wh + gb, &lds[buf][2][lo]);
            copy16_g2l(wl + gb, &lds[buf][3][lo]);
        }
    };

    int cur = 0;
    issue(0, 0);

    for (int k0 = 0; k0 < Ddim; k0 += 32) {
        const bool more = (k0 + 32) < Ddim;
        if (more) issue(cur ^ 1, k0 + 32);    // prefetch next slab
        if (more) wait_async_le<8>();         // retire current slab's 8 copies
        else      wait_async_le<0>();
        __syncthreads();

        // ---- fragments (A: per-lane 16x32 layout; B tile holds B^T rows) ----
        FragBF ah[4], al[4], bh[2], bl[2];
#pragma unroll
        for (int mt = 0; mt < 4; ++mt) {
            int mloc = wm * 64 + mt * 16 + mrow;
            const U4* ph = (const U4*)&lds[cur][0][mloc * LDA + kh * 8];
            const U4* pl = (const U4*)&lds[cur][1][mloc * LDA + kh * 8];
            ah[mt].q[0] = ph[0]; ah[mt].q[1] = ph[2];   // K and K+16 halves
            al[mt].q[0] = pl[0]; al[mt].q[1] = pl[2];
        }
#pragma unroll
        for (int nt = 0; nt < 2; ++nt) {
            int nloc = wn * 32 + nt * 16 + mrow;
            const U4* ph = (const U4*)&lds[cur][2][nloc * LDA + kh * 8];
            const U4* pl = (const U4*)&lds[cur][3][nloc * LDA + kh * 8];
            bh[nt].q[0] = ph[0]; bh[nt].q[1] = ph[2];
            bl[nt].q[0] = pl[0]; bl[nt].q[1] = pl[2];
        }

        // ---- 3-product split-bf16 WMMA ----
#pragma unroll
        for (int mt = 0; mt < 4; ++mt) {
#pragma unroll
            for (int nt = 0; nt < 2; ++nt) {
                v8f a = acc[mt][nt];
                a = __builtin_amdgcn_wmma_f32_16x16x32_bf16(
                        false, ah[mt].v, false, bh[nt].v, (short)0, a, false, false);
                a = __builtin_amdgcn_wmma_f32_16x16x32_bf16(
                        false, ah[mt].v, false, bl[nt].v, (short)0, a, false, false);
                a = __builtin_amdgcn_wmma_f32_16x16x32_bf16(
                        false, al[mt].v, false, bh[nt].v, (short)0, a, false, false);
                acc[mt][nt] = a;
            }
        }
        __syncthreads();   // all reads of lds[cur] done before it is refilled
        cur ^= 1;
    }

    // ---- store z tile (C/D layout: vgpr i -> M=i + 8*(lane/16), N=lane%16) ----
    const int mbase = m0 + wm * 64;
    const int nbase = n0 + wn * 32;
#pragma unroll
    for (int mt = 0; mt < 4; ++mt) {
#pragma unroll
        for (int nt = 0; nt < 2; ++nt) {
#pragma unroll
            for (int i = 0; i < 8; ++i) {
                int m = mbase + mt * 16 + i + (lane >> 4) * 8;
                int n = nbase + nt * 16 + (lane & 15);
                z[(size_t)m * Ddim + n] = acc[mt][nt][i];
            }
        }
    }
}

// =====================================================================
// Kernel 2: per-chunk carries  f[b,d,c,k] = sum_i lam_k^{L-1-i} z[b,cL+i,d]
// =====================================================================
__global__ void __launch_bounds__(256)
chunk_carries(const float* __restrict__ z, const float* __restrict__ lam_logit,
              float* __restrict__ f) {
    const int d = blockIdx.x * 256 + threadIdx.x;
    const int c = blockIdx.y;
    const int b = blockIdx.z;
    float lam[Kdim];
#pragma unroll
    for (int k = 0; k < Kdim; ++k) lam[k] = 1.0f / (1.0f + expf(-lam_logit[k]));

    float m[Kdim] = {0.f, 0.f, 0.f, 0.f};
    const float* zp = z + ((size_t)b * Tdim + (size_t)c * LCHUNK) * Ddim + d;
    for (int i = 0; i < LCHUNK; ++i) {
        float zt = zp[(size_t)i * Ddim];
#pragma unroll
        for (int k = 0; k < Kdim; ++k) m[k] = fmaf(lam[k], m[k], zt);
    }
    size_t o = (((size_t)b * Ddim + d) * NCHUNK + c) * Kdim;
#pragma unroll
    for (int k = 0; k < Kdim; ++k) f[o + k] = m[k];
}

// =====================================================================
// Kernel 3: chunk prefix — incoming state per chunk via lam^L recurrence
// =====================================================================
__global__ void __launch_bounds__(256)
chunk_prefix(const float* __restrict__ lam_logit, const float* __restrict__ f,
             float* __restrict__ m_in) {
    const int idx = blockIdx.x * 256 + threadIdx.x;
    const int b = idx >> 11;
    const int d = idx & (Ddim - 1);
    float lam[Kdim], lamL[Kdim];
#pragma unroll
    for (int k = 0; k < Kdim; ++k) {
        lam[k] = 1.0f / (1.0f + expf(-lam_logit[k]));
        lamL[k] = powf(lam[k], (float)LCHUNK);
    }
    float m[Kdim] = {0.f, 0.f, 0.f, 0.f};
    size_t base = ((size_t)b * Ddim + d) * NCHUNK * Kdim;
    for (int c = 0; c < NCHUNK; ++c) {
#pragma unroll
        for (int k = 0; k < Kdim; ++k) {
            m_in[base + c * Kdim + k] = m[k];
            m[k] = fmaf(lamL[k], m[k], f[base + c * Kdim + k]);
        }
    }
}

// =====================================================================
// Kernel 4: replay scan with carries, write h = z + sum_k eta_k * m_k
// =====================================================================
__global__ void __launch_bounds__(256)
scan_apply(const float* __restrict__ z, const float* __restrict__ lam_logit,
           const float* __restrict__ eta, const float* __restrict__ m_in,
           float* __restrict__ h) {
    const int d = blockIdx.x * 256 + threadIdx.x;
    const int c = blockIdx.y;
    const int b = blockIdx.z;
    float lam[Kdim], et[Kdim];
#pragma unroll
    for (int k = 0; k < Kdim; ++k) {
        lam[k] = 1.0f / (1.0f + expf(-lam_logit[k]));
        et[k] = eta[k];
    }
    float m[Kdim];
    size_t cb = (((size_t)b * Ddim + d) * NCHUNK + c) * Kdim;
#pragma unroll
    for (int k = 0; k < Kdim; ++k) m[k] = m_in[cb + k];

    const size_t off = ((size_t)b * Tdim + (size_t)c * LCHUNK) * Ddim + d;
    const float* zp = z + off;
    float* hp = h + off;
    for (int i = 0; i < LCHUNK; ++i) {
        float zt = zp[(size_t)i * Ddim];
        float s = zt;
#pragma unroll
        for (int k = 0; k < Kdim; ++k) {
            m[k] = fmaf(lam[k], m[k], zt);
            s = fmaf(et[k], m[k], s);
        }
        hp[(size_t)i * Ddim] = s;
    }
}

// =====================================================================
// Kernel 5: in-place LayerNorm over last dim (one block per row)
// =====================================================================
__global__ void __launch_bounds__(256)
layernorm_rows(float* __restrict__ out, const float* __restrict__ gamma,
               const float* __restrict__ beta) {
    const int row = blockIdx.x;
    const int tid = threadIdx.x;
    float* rp = out + (size_t)row * Ddim;

    float v[Ddim / 256];
    float s = 0.f, s2 = 0.f;
#pragma unroll
    for (int j = 0; j < Ddim / 256; ++j) {
        v[j] = rp[j * 256 + tid];
        s += v[j];
        s2 = fmaf(v[j], v[j], s2);
    }
    __shared__ float r1[256], r2[256];
    r1[tid] = s; r2[tid] = s2;
    __syncthreads();
    for (int st = 128; st > 0; st >>= 1) {
        if (tid < st) { r1[tid] += r1[tid + st]; r2[tid] += r2[tid + st]; }
        __syncthreads();
    }
    float mu  = r1[0] * (1.0f / Ddim);
    float var = r2[0] * (1.0f / Ddim) - mu * mu;
    float rs  = rsqrtf(var + LN_EPS);
#pragma unroll
    for (int j = 0; j < Ddim / 256; ++j) {
        int col = j * 256 + tid;
        rp[col] = fmaf(gamma[col], (v[j] - mu) * rs, beta[col]);
    }
}

// =====================================================================
// Host-side launch
// =====================================================================
extern "C" void kernel_launch(void* const* d_in, const int* in_sizes, int n_in,
                              void* d_out, int out_size, void* d_ws, size_t ws_size,
                              hipStream_t stream) {
    const float* x         = (const float*)d_in[0];
    const float* W         = (const float*)d_in[1];
    const float* lam_logit = (const float*)d_in[2];
    const float* eta       = (const float*)d_in[3];
    const float* gamma     = (const float*)d_in[4];
    const float* beta      = (const float*)d_in[5];
    float* out = (float*)d_out;

    const size_t XN = (size_t)Mdim * Ddim;                     // 16.7M
    const size_t WN = (size_t)Ddim * Ddim;                     // 4.2M
    const size_t CE = (size_t)Bdim * Ddim * NCHUNK * Kdim;     // 0.5M

    float* z    = (float*)d_ws;
    float* fcar = z + XN;
    float* minc = fcar + CE;
    unsigned short* xh = (unsigned short*)(minc + CE);
    unsigned short* xl = xh + XN;
    unsigned short* wh = xl + XN;
    unsigned short* wl = wh + WN;

    // 0) split fp32 -> bf16 hi/lo planes (memory-bound, L2-resident after)
    split_bf16<<<(int)(XN / 4 / 256), 256, 0, stream>>>(x, xh, xl, (int)(XN / 4));
    split_bf16<<<(int)(WN / 4 / 256), 256, 0, stream>>>(W, wh, wl, (int)(WN / 4));

    // 1) WMMA GEMM with async global->LDS double buffering
    dim3 gGemm(Ddim / 128, Mdim / 128);
    gemm_split_bf16<<<gGemm, 256, 0, stream>>>(xh, xl, wh, wl, z);

    // 2-4) chunked parallel scan
    dim3 gScan(Ddim / 256, NCHUNK, Bdim);
    chunk_carries<<<gScan, 256, 0, stream>>>(z, lam_logit, fcar);
    chunk_prefix<<<(Bdim * Ddim) / 256, 256, 0, stream>>>(lam_logit, fcar, minc);
    scan_apply<<<gScan, 256, 0, stream>>>(z, lam_logit, eta, minc, out);

    // 5) in-place LayerNorm
    layernorm_rows<<<Mdim, 256, 0, stream>>>(out, gamma, beta);
}
